// PoseRefine_43499428774192
// MI455X (gfx1250) — matrix operand order repaired and checked by reading, not compiled, not run on hardware
//
#include <hip/hip_runtime.h>
#include <hip/hip_bf16.h>

// Problem constants (match reference)
#define B_   16
#define NP_  64
#define J_   17
#define C_   480
#define H_   128
#define W_   128
#define L_   4
#define HW_  (H_ * W_)
#define NROWS (B_ * NP_ * J_)   // 17408 GEMM rows = 544 * 32
#define NS    (B_ * NP_)        // 1024 samples = 32 * 32
#define EPS_  1e-5f

typedef __attribute__((ext_vector_type(16))) _Float16 v16h;
typedef __attribute__((ext_vector_type(8)))  _Float16 v8h;
typedef __attribute__((ext_vector_type(8)))  float    v8f;

// ---------------------------------------------------------------------------
// Weight prep: f32 [K][N] -> f16 transposed [N][K] (so WMMA B-fragments are
// contiguous 16-half runs per lane/column).
// ---------------------------------------------------------------------------
__global__ __launch_bounds__(256) void transpose_cvt_kernel(
    const float* __restrict__ Wsrc, _Float16* __restrict__ Wt, int K, int N) {
  int i = blockIdx.x * 256 + threadIdx.x;
  if (i >= K * N) return;
  int k = i / N, n = i - k * N;
  Wt[(size_t)n * K + k] = (_Float16)Wsrc[i];
}

__global__ __launch_bounds__(256) void cvt_kernel(
    const float* __restrict__ Wsrc, _Float16* __restrict__ Wt, int count) {
  int i = blockIdx.x * 256 + threadIdx.x;
  if (i < count) Wt[i] = (_Float16)Wsrc[i];
}

// ---------------------------------------------------------------------------
// Bilinear sampler: one block per (b,np,j) row, one thread per channel.
// Writes the f16 GEMM A-matrix directly.
// ---------------------------------------------------------------------------
__global__ __launch_bounds__(C_) void sample_kernel(
    const float* __restrict__ feat, const float* __restrict__ coords,
    _Float16* __restrict__ act16) {
  const int row = blockIdx.x;              // (b*NP+np)*J + j
  const int c   = threadIdx.x;             // 0..C-1
  const int b   = row / (NP_ * J_);
  const float x = coords[row * 2 + 0];
  const float y = coords[row * 2 + 1];
  float x0f = fminf(fmaxf(floorf(x), 0.0f), (float)(W_ - 2));
  float y0f = fminf(fmaxf(floorf(y), 0.0f), (float)(H_ - 2));
  const int   i00 = (int)y0f * W_ + (int)x0f;
  const float wx = x - x0f, wy = y - y0f;
  const float* p = feat + ((size_t)b * C_ + c) * HW_ + i00;
  const float f00 = p[0], f01 = p[1], f10 = p[W_], f11 = p[W_ + 1];
  const float v = f00 * (1.0f - wx) * (1.0f - wy) + f01 * wx * (1.0f - wy) +
                  f10 * (1.0f - wx) * wy + f11 * wx * wy;
  act16[(size_t)row * C_ + c] = (_Float16)v;
}

// Center gather: feat_center[b,np,c] = feat[b,c,center_ind[b,np]]
__global__ __launch_bounds__(C_) void center_kernel(
    const float* __restrict__ feat, const int* __restrict__ cind,
    float* __restrict__ featC) {
  const int s = blockIdx.x;                // b*NP+np
  const int c = threadIdx.x;
  const int b = s / NP_;
  const int idx = cind[s];
  featC[(size_t)s * C_ + c] = feat[((size_t)b * C_ + c) * HW_ + idx];
}

// ---------------------------------------------------------------------------
// WMMA GEMM: C[M,N] = A[M,K](f16,row-major) * Bt[N,K](f16, N-major).
// Each wave computes a 32x96 output tile: 2 M sub-tiles x 6 N sub-tiles,
// 12 f32 accumulators. Per 32-K step: 2 A-frags + 6 B-frags feed 12 WMMAs,
// i.e. 0.041 B/FLOP of L2 traffic (2.3x better reuse than a 16x32 tile).
// Fragment layouts used here (contiguous K order, same for A and B):
//   A (16x32 f16): lane<16 holds K k0..k0+15 of row (lane&15);
//                  lane>=16 holds K k0+16..k0+31.
//   B (32x16 f16): lane holds 16 contiguous K of column (lane&15);
//                  lane<16 -> K k0..k0+15, lane>=16 -> K k0+16..k0+31.
//   C/D: VGPR v, lane l -> row m0 + v + 8*(l>=16), col n0 + (l&15).
// Grid is sized so every wave has a valid job -> EXEC all ones at WMMA.
// ---------------------------------------------------------------------------
#define GEMM_WAVES 8
#define TN_SUB 6     // 6 x 16 = 96 columns per wave
__global__ __launch_bounds__(GEMM_WAVES * 32) void gemm_wmma_kernel(
    const _Float16* __restrict__ A, const _Float16* __restrict__ Bt,
    float* __restrict__ Cmat, int M, int N, int K) {
  const int lane = threadIdx.x & 31;
  const int wave = threadIdx.x >> 5;
  const int nstrips = N / 96;                       // 96-wide N strips
  const int job = blockIdx.x * GEMM_WAVES + wave;   // always in range
  const int m0 = (job / nstrips) * 32;
  const int n0 = (job % nstrips) * 96;

  const int halfsel = lane >> 4;                    // 0 | 1
  const int rsel    = lane & 15;

  const _Float16* ap0 = A + (size_t)(m0 + rsel) * K + (halfsel ? 16 : 0);
  const _Float16* ap1 = ap0 + (size_t)16 * K;       // rows m0+16..31
  const _Float16* bp  = Bt + (size_t)(n0 + rsel) * K + (halfsel ? 16 : 0);

  v8f acc[2 * TN_SUB];
#pragma unroll
  for (int i = 0; i < 2 * TN_SUB; ++i) acc[i] = (v8f){};

  for (int k0 = 0; k0 < K; k0 += 32) {
    v16h a0, a1;
    ((v8h*)&a0)[0] = *(const v8h*)(ap0 + k0);       // K k0+{0|16}..+7
    ((v8h*)&a0)[1] = *(const v8h*)(ap0 + k0 + 8);   // K k0+{8|24}..+7
    ((v8h*)&a1)[0] = *(const v8h*)(ap1 + k0);
    ((v8h*)&a1)[1] = *(const v8h*)(ap1 + k0 + 8);
#pragma unroll
    for (int t = 0; t < TN_SUB; ++t) {
      const v16h b = *(const v16h*)(bp + (size_t)(16 * t) * K + k0);
      acc[t] = __builtin_amdgcn_wmma_f32_16x16x32_f16(
          false, a0, false, b, (short)0, acc[t], false, false);
      acc[TN_SUB + t] = __builtin_amdgcn_wmma_f32_16x16x32_f16(
          false, a1, false, b, (short)0, acc[TN_SUB + t], false, false);
    }
  }

  float* crow0 = Cmat + (size_t)(m0 + (halfsel ? 8 : 0)) * N + n0 + rsel;
  float* crow1 = crow0 + (size_t)16 * N;
#pragma unroll
  for (int t = 0; t < TN_SUB; ++t) {
#pragma unroll
    for (int v = 0; v < 8; ++v) {
      crow0[(size_t)v * N + 16 * t] = acc[t][v];
      crow1[(size_t)v * N + 16 * t] = acc[TN_SUB + t][v];
    }
  }
}

// ---------------------------------------------------------------------------
// Fused graph-conv epilogue: z[n,j,c] = relu(BN(sum_k adj[j,k]*y[n,k,c] + b[c]))
// (+ residual), writing both f32 state and the f16 A-matrix for the next GEMM.
// One block per sample n; y[n] (17x480 f32 = 32.6KB) staged in LDS.
// ---------------------------------------------------------------------------
__global__ __launch_bounds__(256) void adj_bn_relu_kernel(
    const float* __restrict__ y, const float* __restrict__ adj,
    const float* __restrict__ bias, const float* __restrict__ bn,
    const float* __restrict__ res, float* __restrict__ hOut,
    _Float16* __restrict__ act16) {
  __shared__ float ybuf[J_ * C_];
  __shared__ float adjs[J_ * J_];
  const int n = blockIdx.x;
  const float* yn = y + (size_t)n * (J_ * C_);
  for (int i = threadIdx.x; i < J_ * C_; i += 256) ybuf[i] = yn[i];
  for (int i = threadIdx.x; i < J_ * J_; i += 256) adjs[i] = adj[i];
  __syncthreads();
  for (int i = threadIdx.x; i < J_ * C_; i += 256) {
    const int j = i / C_;
    const int c = i - j * C_;
    float s = 0.0f;
#pragma unroll
    for (int k = 0; k < J_; ++k) s += adjs[j * J_ + k] * ybuf[k * C_ + c];
    s += bias[c];
    float v = (s - bn[2 * C_ + c]) * rsqrtf(bn[3 * C_ + c] + EPS_) * bn[c] +
              bn[C_ + c];
    v = fmaxf(v, 0.0f);
    if (res) v += res[(size_t)n * (J_ * C_) + i];
    hOut[(size_t)n * (J_ * C_) + i]  = v;
    act16[(size_t)n * (J_ * C_) + i] = (_Float16)v;
  }
}

// Pool over joints + concat with center features -> f16 GEMM input [NS, 3C]
__global__ __launch_bounds__(C_) void pool_concat_kernel(
    const float* __restrict__ h, const float* __restrict__ featC,
    _Float16* __restrict__ feats16) {
  const int n = blockIdx.x;
  const int c = threadIdx.x;
  const float* hn = h + (size_t)n * (J_ * C_) + c;
  float mean = 0.0f, mx = -INFINITY;
#pragma unroll
  for (int j = 0; j < J_; ++j) {
    const float v = hn[(size_t)j * C_];
    mean += v;
    mx = fmaxf(mx, v);
  }
  mean *= (1.0f / (float)J_);
  _Float16* f = feats16 + (size_t)n * (3 * C_);
  f[c]           = (_Float16)mean;
  f[C_ + c]      = (_Float16)mx;
  f[2 * C_ + c]  = (_Float16)featC[(size_t)n * C_ + c];
}

// Prediction-head epilogue: p = relu(BN(y + bp1))
__global__ __launch_bounds__(256) void pred_bn_relu_kernel(
    const float* __restrict__ y, const float* __restrict__ bias,
    const float* __restrict__ bn, float* __restrict__ p) {
  const int i = blockIdx.x * 256 + threadIdx.x;
  if (i >= NS * C_) return;
  const int c = i % C_;
  const float s = y[i] + bias[c];
  const float v = (s - bn[2 * C_ + c]) * rsqrtf(bn[3 * C_ + c] + EPS_) * bn[c] +
                  bn[C_ + c];
  p[i] = fmaxf(v, 0.0f);
}

// Final tiny layer: out[n,o] = sum_c p[n,c]*Wp2[o,c] + bp2[o]  (N=34, VALU)
__global__ __launch_bounds__(64) void final_kernel(
    const float* __restrict__ p, const float* __restrict__ W2,
    const float* __restrict__ b2, float* __restrict__ out) {
  const int n = blockIdx.x;
  const int o = threadIdx.x;
  if (o >= 2 * J_) return;
  const float* pn = p + (size_t)n * C_;
  const float* w  = W2 + (size_t)o * C_;
  float s = b2[o];
  for (int c = 0; c < C_; ++c) s += pn[c] * w[c];
  out[(size_t)n * (2 * J_) + o] = s;
}

// ---------------------------------------------------------------------------
extern "C" void kernel_launch(void* const* d_in, const int* in_sizes, int n_in,
                              void* d_out, int out_size, void* d_ws, size_t ws_size,
                              hipStream_t stream) {
  const float* features  = (const float*)d_in[0];
  const float* coords    = (const float*)d_in[1];
  const int*   centerInd = (const int*)d_in[2];
  const float* adj       = (const float*)d_in[3];
  const float* Wh        = (const float*)d_in[4];
  const float* bh        = (const float*)d_in[5];
  const float* bnHead    = (const float*)d_in[6];
  const float* Wres      = (const float*)d_in[7];
  const float* bres      = (const float*)d_in[8];
  const float* bnRes     = (const float*)d_in[9];
  const float* Wp1       = (const float*)d_in[10];
  const float* bp1       = (const float*)d_in[11];
  const float* bnPred    = (const float*)d_in[12];
  const float* Wp2       = (const float*)d_in[13];
  const float* bp2       = (const float*)d_in[14];
  float* out = (float*)d_out;

  // Workspace carve-up (256B aligned)
  char* ws = (char*)d_ws;
  size_t off = 0;
  auto carve = [&](size_t bytes) {
    char* p = ws + off;
    off = (off + bytes + 255) & ~(size_t)255;
    return p;
  };
  _Float16* act16   = (_Float16*)carve((size_t)NROWS * C_ * 2);     // GEMM A (f16)
  float*    y32     = (float*)carve((size_t)NROWS * C_ * 4);        // GEMM out
  float*    hP      = (float*)carve((size_t)NROWS * C_ * 4);        // h state / residual
  float*    hQ      = (float*)carve((size_t)NROWS * C_ * 4);        // intra-block temp
  float*    featC   = (float*)carve((size_t)NS * C_ * 4);
  _Float16* feats16 = (_Float16*)carve((size_t)NS * 3 * C_ * 2);
  _Float16* WhT     = (_Float16*)carve((size_t)C_ * C_ * 2);
  _Float16* WresT   = (_Float16*)carve((size_t)2 * L_ * C_ * C_ * 2);
  _Float16* Wp1T    = (_Float16*)carve((size_t)C_ * 3 * C_ * 2);
  float*    p32     = hQ;   // reuse (1024*480 << NROWS*C)
  float*    y2      = y32;  // reuse

  // ---- 1. Weight prep (f32 -> f16, transpose [K][N] -> [N][K]) ----
  const int wgridCC = (C_ * C_ + 255) / 256;
  transpose_cvt_kernel<<<wgridCC, 256, 0, stream>>>(Wh, WhT, C_, C_);
  for (int i = 0; i < 2 * L_; ++i) {
    transpose_cvt_kernel<<<wgridCC, 256, 0, stream>>>(
        Wres + (size_t)i * C_ * C_, WresT + (size_t)i * C_ * C_, C_, C_);
  }
  // Wp1 is [C_out, 3C] row-major == already [N][K]: plain convert
  cvt_kernel<<<(C_ * 3 * C_ + 255) / 256, 256, 0, stream>>>(Wp1, Wp1T, C_ * 3 * C_);

  // ---- 2. Sampling ----
  sample_kernel<<<NROWS, C_, 0, stream>>>(features, coords, act16);
  center_kernel<<<NS, C_, 0, stream>>>(features, centerInd, featC);

  // ---- 3. Head + residual graph-conv blocks ----
  const int jobsMain   = (NROWS / 32) * (C_ / 96);       // 544*5 = 2720
  const int blocksMain = jobsMain / GEMM_WAVES;          // 340 (exact)

  // head: h = g(featJ, Wh, bh, bnHead)
  gemm_wmma_kernel<<<blocksMain, GEMM_WAVES * 32, 0, stream>>>(
      act16, WhT, y32, NROWS, C_, C_);
  adj_bn_relu_kernel<<<NS, 256, 0, stream>>>(y32, adj, bh, bnHead,
                                             nullptr, hP, act16);
  for (int i = 0; i < L_; ++i) {
    // conv1: hQ = g(hP) ; hP kept as residual
    gemm_wmma_kernel<<<blocksMain, GEMM_WAVES * 32, 0, stream>>>(
        act16, WresT + (size_t)(2 * i) * C_ * C_, y32, NROWS, C_, C_);
    adj_bn_relu_kernel<<<NS, 256, 0, stream>>>(
        y32, adj, bres + (size_t)(2 * i) * C_, bnRes + (size_t)(2 * i) * 4 * C_,
        nullptr, hQ, act16);
    // conv2: hP = hP + g(hQ)
    gemm_wmma_kernel<<<blocksMain, GEMM_WAVES * 32, 0, stream>>>(
        act16, WresT + (size_t)(2 * i + 1) * C_ * C_, y32, NROWS, C_, C_);
    adj_bn_relu_kernel<<<NS, 256, 0, stream>>>(
        y32, adj, bres + (size_t)(2 * i + 1) * C_,
        bnRes + (size_t)(2 * i + 1) * 4 * C_, hP, hP, act16);
  }

  // ---- 4. Pool/concat + prediction head ----
  pool_concat_kernel<<<NS, C_, 0, stream>>>(hP, featC, feats16);
  const int jobsPred   = (NS / 32) * (C_ / 96);          // 32*5 = 160
  const int blocksPred = jobsPred / GEMM_WAVES;          // 20 (exact)
  gemm_wmma_kernel<<<blocksPred, GEMM_WAVES * 32, 0, stream>>>(
      feats16, Wp1T, y2, NS, C_, 3 * C_);
  pred_bn_relu_kernel<<<(NS * C_ + 255) / 256, 256, 0, stream>>>(
      y2, bp1, bnPred, p32);

  // ---- 5. Final tiny layer ----
  final_kernel<<<NS, 64, 0, stream>>>(p32, Wp2, bp2, out);
}
